// LlamaAttention_41180146434065
// MI455X (gfx1250) — matrix operand order repaired and checked by reading, not compiled, not run on hardware
//
#include <hip/hip_runtime.h>
#include <cstdint>
#include <cstddef>

// ---------- types ----------
typedef unsigned int   u32;
typedef unsigned short u16;
typedef u32    u32x4   __attribute__((ext_vector_type(4)));
typedef __bf16 bf16x16 __attribute__((ext_vector_type(16)));
typedef float  f32x8   __attribute__((ext_vector_type(8)));

union V16 { bf16x16 v; u32x4 q[2]; u32 u[8]; };

// ---------- problem constants ----------
constexpr int B_   = 2;
constexpr int T_   = 2048;
constexpr int HID_ = 2048;
constexpr int NH_  = 32;
constexpr int NKV_ = 8;
constexpr int HD_  = 64;

// ---------- helpers ----------
__device__ inline u16 f2bf(float f) {           // RNE f32 -> bf16
  u32 u = __float_as_uint(f);
  u32 r = (u + 0x7fffu + ((u >> 16) & 1u)) >> 16;
  return (u16)r;
}
__device__ inline float bf2f(u16 h) { return __uint_as_float(((u32)h) << 16); }

__device__ inline f32x8 zero8() {
  f32x8 z;
#pragma unroll
  for (int r = 0; r < 8; r++) z[r] = 0.f;
  return z;
}

__device__ inline f32x8 wmma_bf16(bf16x16 a, bf16x16 b, f32x8 c) {
  // D = A(16x32 bf16) * B(32x16 bf16) + C(16x16 f32)
  return __builtin_amdgcn_wmma_f32_16x16x32_bf16(
      /*neg_a=*/false, a, /*neg_b=*/false, b,
      /*c_mod=*/(short)0, c, /*reuse_a=*/false, /*reuse_b=*/false);
}

// ---------- kernel: f32 -> bf16 convert ----------
__global__ void k_f32_to_bf16(const float* __restrict__ in, u16* __restrict__ out, int n) {
  int i = blockIdx.x * blockDim.x + threadIdx.x;
  if (i < n) out[i] = f2bf(in[i]);
}

// ---------- kernel: generic NT GEMM, C = A(MxK) * W(NxK)^T ----------
// A,W bf16 row-major. Each wave computes a 64x64 C block (16 WMMA tiles).
// A-layout per lane:  m = lane%16, element i -> k = k0 + (i%8) + (i/8)*16 + (lane/16)*8
// B-layout per lane:  n = lane%16, element i -> k = k0 + (i%8) + (i/8)*16 + (lane/16)*8
template <int BF16OUT>
__global__ __launch_bounds__(128) void k_gemm_nt(const u16* __restrict__ A,
                                                 const u16* __restrict__ W,
                                                 void* __restrict__ Cptr,
                                                 int M, int N, int K) {
  const int lane = threadIdx.x & 31;
  const int wave = threadIdx.x >> 5;
  const int l16  = lane & 15;
  const int hf   = lane >> 4;
  const int wm = blockIdx.y * 128 + (wave >> 1) * 64;
  const int wn = blockIdx.x * 128 + (wave & 1) * 64;

  f32x8 c[4][4];
#pragma unroll
  for (int i = 0; i < 4; i++)
#pragma unroll
    for (int j = 0; j < 4; j++) c[i][j] = zero8();

  for (int k0 = 0; k0 < K; k0 += 32) {
    V16 a[4];
#pragma unroll
    for (int mt = 0; mt < 4; mt++) {
      const u16* ap = A + (size_t)(wm + mt * 16 + l16) * K + k0 + hf * 8;
      a[mt].q[0] = *(const u32x4*)(ap);
      a[mt].q[1] = *(const u32x4*)(ap + 16);
    }
#pragma unroll
    for (int nt = 0; nt < 4; nt++) {
      V16 bm;
      const u16* wp = W + (size_t)(wn + nt * 16 + l16) * K + k0 + hf * 8;
      bm.q[0] = *(const u32x4*)(wp);
      bm.q[1] = *(const u32x4*)(wp + 16);
#pragma unroll
      for (int mt = 0; mt < 4; mt++) c[mt][nt] = wmma_bf16(a[mt].v, bm.v, c[mt][nt]);
    }
  }

#pragma unroll
  for (int mt = 0; mt < 4; mt++)
#pragma unroll
    for (int nt = 0; nt < 4; nt++)
#pragma unroll
      for (int r = 0; r < 8; r++) {
        int gm = wm + mt * 16 + hf * 8 + r;   // C layout: m = r + 8*(lane/16)
        int gn = wn + nt * 16 + l16;          //           n = lane%16
        float val = c[mt][nt][r];
        if (BF16OUT) ((u16*)Cptr)[(size_t)gm * N + gn] = f2bf(val);
        else         ((float*)Cptr)[(size_t)gm * N + gn] = val;
      }
}

// ---------- kernel: RoPE + pack (b,t,h,d) -> (b,h,t,d) ----------
__global__ void k_rope_pack(const u16* __restrict__ raw, u16* __restrict__ packed,
                            int heads, float scale, int n) {
  int i = blockIdx.x * blockDim.x + threadIdx.x;
  if (i >= n) return;
  int d    = i & 63;
  int rest = i >> 6;
  int h    = rest % heads;
  int bt   = rest / heads;
  int t    = bt % T_;
  int b    = bt / T_;
  int j    = d & 31;
  // inv_freq = 10000^(-j/32) = 2^(-j * log2(10000)/32)
  float ang = (float)t * exp2f(-(float)j * 0.41524138f);
  float s, cosv;
  __sincosf(ang, &s, &cosv);
  float v  = bf2f(raw[i]);
  int   pi = (i & ~63) | ((d < 32) ? (d + 32) : (d - 32));
  float pv = bf2f(raw[pi]);
  float res = (d < 32) ? (v * cosv - pv * s) : (v * cosv + pv * s);
  packed[((size_t)(b * heads + h) * T_ + t) * HD_ + d] = f2bf(res * scale);
}

// ---------- kernel: pack V (b,t,h,d) -> (b,h,d,t)  (transposed for PV B-operand) ----------
__global__ void k_pack_vt(const u16* __restrict__ raw, u16* __restrict__ vt, int n) {
  int i = blockIdx.x * blockDim.x + threadIdx.x;
  if (i >= n) return;
  int d    = i & 63;
  int rest = i >> 6;
  int h    = rest % NKV_;
  int bt   = rest / NKV_;
  int t    = bt % T_;
  int b    = bt / T_;
  vt[((size_t)(b * NKV_ + h) * HD_ + d) * T_ + t] = raw[i];
}

// ---------- kernel: flash attention, one wave per (b, h, 16-query tile) ----------
__global__ __launch_bounds__(32) void k_attn(const u16* __restrict__ Qp,
                                             const u16* __restrict__ Kp,
                                             const u16* __restrict__ Vt,
                                             u16* __restrict__ Ao) {
  __shared__ u16 plds[16][32];
  const int lane = threadIdx.x;
  const int l16  = lane & 15;
  const int hf   = lane >> 4;
  const int q0 = blockIdx.x * 16;
  const int h  = blockIdx.y;
  const int b  = blockIdx.z;
  const int hk = h >> 2;  // NH/NKV = 4

  // Q tile 16x64 in A-layout (two k-chunks of 32), scale 1/sqrt(HD) folded at pack
  const u16* qb = Qp + ((size_t)(b * NH_ + h) * T_ + q0 + l16) * HD_;
  V16 aq0, aq1;
  aq0.q[0] = *(const u32x4*)(qb + hf * 8);
  aq0.q[1] = *(const u32x4*)(qb + hf * 8 + 16);
  aq1.q[0] = *(const u32x4*)(qb + 32 + hf * 8);
  aq1.q[1] = *(const u32x4*)(qb + 32 + hf * 8 + 16);

  const u16* kb_base = Kp + (size_t)(b * NKV_ + hk) * T_ * HD_;
  const u16* vb_base = Vt + (size_t)(b * NKV_ + hk) * HD_ * T_;

  f32x8 o[4];
#pragma unroll
  for (int t = 0; t < 4; t++) o[t] = zero8();
  float mr[8], lr[8];
#pragma unroll
  for (int r = 0; r < 8; r++) { mr[r] = -3.0e38f; lr[r] = 0.f; }

  for (int kb = 0; kb < q0 + 16; kb += 32) {
    // S = Q Kt : two 16x16 tiles (keys kb..+15, kb+16..+31), k-dim = HD = 2 WMMA steps
    f32x8 s0 = zero8(), s1 = zero8();
    const u16* kr0 = kb_base + (size_t)(kb + l16) * HD_;
    const u16* kr1 = kr0 + 16 * HD_;
    V16 bk;
    bk.q[0] = *(const u32x4*)(kr0 + hf * 8);
    bk.q[1] = *(const u32x4*)(kr0 + hf * 8 + 16);
    s0 = wmma_bf16(aq0.v, bk.v, s0);
    bk.q[0] = *(const u32x4*)(kr0 + 32 + hf * 8);
    bk.q[1] = *(const u32x4*)(kr0 + 32 + hf * 8 + 16);
    s0 = wmma_bf16(aq1.v, bk.v, s0);
    bk.q[0] = *(const u32x4*)(kr1 + hf * 8);
    bk.q[1] = *(const u32x4*)(kr1 + hf * 8 + 16);
    s1 = wmma_bf16(aq0.v, bk.v, s1);
    bk.q[0] = *(const u32x4*)(kr1 + 32 + hf * 8);
    bk.q[1] = *(const u32x4*)(kr1 + 32 + hf * 8 + 16);
    s1 = wmma_bf16(aq1.v, bk.v, s1);

    // causal mask (C layout: row = q0 + hf*8 + r, key = kb + l16 (+16))
    const int key0 = kb + l16;
    const int key1 = key0 + 16;
    const int rowb = q0 + hf * 8;
#pragma unroll
    for (int r = 0; r < 8; r++) {
      int row = rowb + r;
      if (key0 > row) s0[r] = -3.0e38f;
      if (key1 > row) s1[r] = -3.0e38f;
    }

    // online softmax: row-max / row-sum across the 16 lanes of each half
    float tmax[8];
#pragma unroll
    for (int r = 0; r < 8; r++) tmax[r] = fmaxf(s0[r], s1[r]);
#pragma unroll
    for (int msk = 1; msk <= 8; msk <<= 1)
#pragma unroll
      for (int r = 0; r < 8; r++) tmax[r] = fmaxf(tmax[r], __shfl_xor(tmax[r], msk, 32));

    float alpha[8];
#pragma unroll
    for (int r = 0; r < 8; r++) {
      float mn = fmaxf(mr[r], tmax[r]);
      alpha[r] = __expf(mr[r] - mn);
      mr[r] = mn;
    }
    float p0[8], p1[8], rs[8];
#pragma unroll
    for (int r = 0; r < 8; r++) {
      p0[r] = __expf(s0[r] - mr[r]);
      p1[r] = __expf(s1[r] - mr[r]);
      rs[r] = p0[r] + p1[r];
    }
#pragma unroll
    for (int msk = 1; msk <= 8; msk <<= 1)
#pragma unroll
      for (int r = 0; r < 8; r++) rs[r] += __shfl_xor(rs[r], msk, 32);
#pragma unroll
    for (int r = 0; r < 8; r++) lr[r] = lr[r] * alpha[r] + rs[r];
#pragma unroll
    for (int t = 0; t < 4; t++)
#pragma unroll
      for (int r = 0; r < 8; r++) o[t][r] *= alpha[r];

    // P (C layout) -> LDS -> A layout
    __syncthreads();
#pragma unroll
    for (int r = 0; r < 8; r++) {
      plds[hf * 8 + r][l16]      = f2bf(p0[r]);
      plds[hf * 8 + r][16 + l16] = f2bf(p1[r]);
    }
    __syncthreads();
    V16 pa;
#pragma unroll
    for (int j = 0; j < 8; j++) {
      int k2 = ((j < 4) ? 2 * j : 2 * j + 8) + hf * 8;  // A-layout k for elem pair (2j,2j+1)
      pa.u[j] = *(const u32*)&plds[l16][k2];
    }

    // O += P(16x32) * V(32x64): 4 WMMAs, B from transposed V (keys contiguous)
#pragma unroll
    for (int t = 0; t < 4; t++) {
      const u16* vr = vb_base + (size_t)(t * 16 + l16) * T_ + kb + hf * 8;
      V16 bv;
      bv.q[0] = *(const u32x4*)(vr);
      bv.q[1] = *(const u32x4*)(vr + 16);
      o[t] = wmma_bf16(pa.v, bv.v, o[t]);
    }
  }

  float inv[8];
#pragma unroll
  for (int r = 0; r < 8; r++) inv[r] = 1.0f / lr[r];
#pragma unroll
  for (int t = 0; t < 4; t++)
#pragma unroll
    for (int r = 0; r < 8; r++) {
      int gm = q0 + hf * 8 + r;
      int gn = h * HD_ + t * 16 + l16;
      Ao[(size_t)(b * T_ + gm) * (NH_ * HD_) + gn] = f2bf(o[t][r] * inv[r]);
    }
}

// ---------- host launcher ----------
extern "C" void kernel_launch(void* const* d_in, const int* in_sizes, int n_in,
                              void* d_out, int out_size, void* d_ws, size_t ws_size,
                              hipStream_t stream) {
  const float* x  = (const float*)d_in[0];
  // d_in[1] = attention_mask (causal, recomputed analytically -> ignored)
  const float* qw = (const float*)d_in[2];
  const float* kw = (const float*)d_in[3];
  const float* vw = (const float*)d_in[4];
  const float* ow = (const float*)d_in[5];

  const size_t nX  = (size_t)B_ * T_ * HID_;       // 8388608
  const size_t nQw = (size_t)NH_ * HD_ * HID_;     // 4194304
  const size_t nKw = (size_t)NKV_ * HD_ * HID_;    // 1048576
  const size_t nVw = nKw;
  const size_t nOw = (size_t)HID_ * NH_ * HD_;     // 4194304
  const size_t nQ  = (size_t)B_ * T_ * NH_ * HD_;  // 8388608
  const size_t nK  = (size_t)B_ * T_ * NKV_ * HD_; // 2097152
  const size_t nV  = nK;

  char* ws = (char*)d_ws;
  size_t off = 0;
  auto take = [&](size_t bytes) -> char* {
    char* p = ws + off;
    off += (bytes + 255) & ~(size_t)255;
    return p;
  };
  u16* xb    = (u16*)take(nX * 2);    // x bf16; later reused as Qp (packed Q)
  u16* qwb   = (u16*)take(nQw * 2);
  u16* kwb   = (u16*)take(nKw * 2);
  u16* vwb   = (u16*)take(nVw * 2);
  u16* owb   = (u16*)take(nOw * 2);
  u16* q_raw = (u16*)take(nQ * 2);    // later reused as attn_out
  u16* k_raw = (u16*)take(nK * 2);
  u16* v_raw = (u16*)take(nV * 2);
  u16* Kp    = (u16*)take(nK * 2);
  u16* Vtr   = (u16*)take(nV * 2);
  if (off > ws_size) return;  // workspace too small: bail (deterministically)

  u16* Qp = xb;      // safe alias: xb dead after the three QKV GEMMs
  u16* Ao = q_raw;   // safe alias: q_raw dead after RoPE pack

  const int CT = 256;
  // 1) convert inputs/weights to bf16
  k_f32_to_bf16<<<(int)((nX  + CT - 1) / CT), CT, 0, stream>>>(x,  xb,  (int)nX);
  k_f32_to_bf16<<<(int)((nQw + CT - 1) / CT), CT, 0, stream>>>(qw, qwb, (int)nQw);
  k_f32_to_bf16<<<(int)((nKw + CT - 1) / CT), CT, 0, stream>>>(kw, kwb, (int)nKw);
  k_f32_to_bf16<<<(int)((nVw + CT - 1) / CT), CT, 0, stream>>>(vw, vwb, (int)nVw);
  k_f32_to_bf16<<<(int)((nOw + CT - 1) / CT), CT, 0, stream>>>(ow, owb, (int)nOw);

  // 2) QKV projections (WMMA GEMM, bf16 out)
  const int M = B_ * T_;  // 4096
  k_gemm_nt<1><<<dim3(2048 / 128, M / 128), 128, 0, stream>>>(xb, qwb, q_raw, M, 2048, HID_);
  k_gemm_nt<1><<<dim3(512 / 128,  M / 128), 128, 0, stream>>>(xb, kwb, k_raw, M, 512,  HID_);
  k_gemm_nt<1><<<dim3(512 / 128,  M / 128), 128, 0, stream>>>(xb, vwb, v_raw, M, 512,  HID_);

  // 3) RoPE + repack (Q gets 1/sqrt(HD) folded in); V transposed for PV GEMM
  k_rope_pack<<<(int)((nQ + CT - 1) / CT), CT, 0, stream>>>(q_raw, Qp, NH_,  0.125f, (int)nQ);
  k_rope_pack<<<(int)((nK + CT - 1) / CT), CT, 0, stream>>>(k_raw, Kp, NKV_, 1.0f,   (int)nK);
  k_pack_vt  <<<(int)((nV + CT - 1) / CT), CT, 0, stream>>>(v_raw, Vtr, (int)nV);

  // 4) causal flash attention (one wave per 16 query rows)
  k_attn<<<dim3(T_ / 16, NH_, B_), 32, 0, stream>>>(Qp, Kp, Vtr, Ao);

  // 5) output projection (f32 out)
  k_gemm_nt<0><<<dim3(2048 / 128, M / 128), 128, 0, stream>>>(Ao, owb, d_out, M, 2048, HID_);
}
